// InfoNCELoss_4045859193667
// MI455X (gfx1250) — matrix operand order repaired and checked by reading, not compiled, not run on hardware
//
#include <hip/hip_runtime.h>
#include <hip/hip_bf16.h>

// ---------------------------------------------------------------------------
// InfoNCE / supervised-contrastive loss, fused on CDNA5 (gfx1250, wave32).
//   B=8, N=2048, D=256, T=0.1
//   sim = normalize(emb) @ normalize(emb)^T / T   (never materialized)
//   loss = mean over valid anchors of -log( sum_pos exp(sim) / sum_all exp(sim) )
// GEMM runs on v_wmma_f32_16x16x32_bf16; epilogue fused in registers.
// B fragments are double-buffered in VGPRs so ds_load latency hides behind WMMA.
// ---------------------------------------------------------------------------

typedef __attribute__((ext_vector_type(16))) __bf16 v16bf;
typedef __attribute__((ext_vector_type(8)))  float  v8f;

union Frag {
    v16bf v;
    uint4 q[2];
};

#define N_   2048
#define D_   256
#define CT   64            // columns staged per LDS tile
#define CTP  72            // padded LDS stride; 72*2B = 144 = 16*9 -> rows 16B aligned
#define INV_T 10.0f        // 1 / 0.1

static __device__ __forceinline__ unsigned short f2bf(float f) {
    unsigned int u = __float_as_uint(f);
    unsigned int r = (u + 0x7FFFu + ((u >> 16) & 1u)) >> 16;   // RNE
    return (unsigned short)r;
}

// --- kernel 1: L2-normalize rows, convert to bf16 (one wave per row) -------
__global__ __launch_bounds__(256)
void nrm_kernel(const float* __restrict__ emb, unsigned short* __restrict__ zb,
                int nRows) {
    int row  = (int)((blockIdx.x * blockDim.x + threadIdx.x) >> 5);
    int lane = (int)(threadIdx.x & 31);
    if (row >= nRows) return;

    const float4* src = (const float4*)(emb + (size_t)row * D_);
    float4 a = src[lane];        // elems 4*lane     .. +3
    float4 b = src[lane + 32];   // elems 128+4*lane .. +3

    float ss = a.x*a.x + a.y*a.y + a.z*a.z + a.w*a.w
             + b.x*b.x + b.y*b.y + b.z*b.z + b.w*b.w;
#pragma unroll
    for (int m = 16; m >= 1; m >>= 1) ss += __shfl_xor(ss, m, 32);

    float inv = 1.0f / fmaxf(sqrtf(ss), 1e-12f);

    unsigned short* dst = zb + (size_t)row * D_;
    uint2 p0, p1;
    p0.x = (unsigned)f2bf(a.x*inv) | ((unsigned)f2bf(a.y*inv) << 16);
    p0.y = (unsigned)f2bf(a.z*inv) | ((unsigned)f2bf(a.w*inv) << 16);
    p1.x = (unsigned)f2bf(b.x*inv) | ((unsigned)f2bf(b.y*inv) << 16);
    p1.y = (unsigned)f2bf(b.z*inv) | ((unsigned)f2bf(b.w*inv) << 16);
    ((uint2*)(dst + 4*lane))[0]       = p0;
    ((uint2*)(dst + 128 + 4*lane))[0] = p1;
}

// --- kernel 2: fused WMMA similarity + masked softmax-sum epilogue ---------
// grid = B * (N/128); block = 256 threads (8 waves); wave handles 16 rows.
__global__ __launch_bounds__(256)
void ctr_kernel(const unsigned short* __restrict__ zb,
                const int*            __restrict__ groups,
                const unsigned char*  __restrict__ mask,
                float*                __restrict__ accum /* [0]=loss sum, [1]=count */) {
    __shared__ unsigned short ldsB[D_ * CTP];   // transposed column tile [k][c]
    __shared__ float colMask[CT];
    __shared__ int   colGroup[CT];

    const int b        = (int)blockIdx.x >> 4;        // N/128 = 16 row tiles
    const int rowTile  = (int)blockIdx.x & 15;
    const int tid      = (int)threadIdx.x;
    const int wave     = tid >> 5;
    const int lane     = tid & 31;
    const int laneRow  = lane & 15;
    const int hi       = lane >> 4;                   // half-wave select
    const int mhalf    = hi * 8;
    const int rowbase  = rowTile * 128 + wave * 16;   // this wave's 16 anchors

    const unsigned short* zbB  = zb     + (size_t)b * N_ * D_;
    const int*            grpB = groups + (size_t)b * N_;
    const unsigned char*  mskB = mask   + (size_t)b * N_;

    // ---- A fragments for this wave's 16 rows, all 8 K-steps, kept resident
    Frag af[8];
    {
        const uint4* base = (const uint4*)(zbB + (size_t)(rowbase + laneRow) * D_);
#pragma unroll
        for (int s = 0; s < 8; ++s) {
            af[s].q[0] = base[4*s + hi];        // K = 32s+8hi    .. +7
            af[s].q[1] = base[4*s + hi + 2];    // K = 32s+8hi+16 .. +23
        }
    }

    // ---- per-lane row metadata (rows mhalf..mhalf+7 of this wave)
    float mrow[8];
    int   grow[8];
#pragma unroll
    for (int r = 0; r < 8; ++r) {
        int i   = rowbase + mhalf + r;
        mrow[r] = (float)mskB[i];
        grow[r] = grpB[i];
    }

    float dsum[8], nsum[8];
#pragma unroll
    for (int r = 0; r < 8; ++r) { dsum[r] = 0.0f; nsum[r] = 0.0f; }

    // staging decomposition: each thread owns one 8(c) x 8(k) block
    const int cb = (tid & 7) * 8;    // column block base   (0..56)
    const int kb = (tid >> 3) * 8;   // K block base        (0..248)

    // ---- sweep all columns in 64-wide LDS-staged tiles
    for (int ct = 0; ct < N_ / CT; ++ct) {
        const int colbase = ct * CT;
        __syncthreads();
        {   // stage transposed tile: ldsB[k*CTP + c] = z[colbase+c][k]
            // 8x8 register-block transpose -> vector ds_store_b128 writes
            const unsigned short* srcp = zbB + (size_t)(colbase + cb) * D_ + kb;
            uint4 r[8];
#pragma unroll
            for (int i = 0; i < 8; ++i)
                r[i] = *(const uint4*)(srcp + (size_t)i * D_);

#pragma unroll
            for (int k = 0; k < 8; ++k) {
                const int d  = k >> 1;
                const int sh = (k & 1) * 16;
                unsigned e0 = (((const unsigned*)&r[0])[d] >> sh) & 0xffffu;
                unsigned e1 = (((const unsigned*)&r[1])[d] >> sh) & 0xffffu;
                unsigned e2 = (((const unsigned*)&r[2])[d] >> sh) & 0xffffu;
                unsigned e3 = (((const unsigned*)&r[3])[d] >> sh) & 0xffffu;
                unsigned e4 = (((const unsigned*)&r[4])[d] >> sh) & 0xffffu;
                unsigned e5 = (((const unsigned*)&r[5])[d] >> sh) & 0xffffu;
                unsigned e6 = (((const unsigned*)&r[6])[d] >> sh) & 0xffffu;
                unsigned e7 = (((const unsigned*)&r[7])[d] >> sh) & 0xffffu;
                uint4 out;
                out.x = e0 | (e1 << 16);
                out.y = e2 | (e3 << 16);
                out.z = e4 | (e5 << 16);
                out.w = e6 | (e7 << 16);
                *(uint4*)&ldsB[(kb + k) * CTP + cb] = out;   // 16B aligned
            }
            if (tid < CT) {
                colMask[tid]  = (float)mskB[colbase + tid];
                colGroup[tid] = grpB[colbase + tid];
            }
        }
        __syncthreads();

        // prefetch next column tile while we compute (global_prefetch_b8)
        if (ct + 1 < N_ / CT) {
            const unsigned short* nxt =
                zbB + (size_t)(colbase + CT + (tid & 63)) * D_ + (tid >> 6) * 64;
            __builtin_prefetch(nxt, 0, 1);
        }

#pragma unroll
        for (int j = 0; j < 4; ++j) {                 // four 16-col subtiles
            const float mJ    = colMask[j * 16 + laneRow];
            const int   gJ    = colGroup[j * 16 + laneRow];
            const int   jglob = colbase + j * 16 + laneRow;

            // ---- K loop with B-fragment double buffering in registers ----
            Frag bfr[2];
            {
                const unsigned short* p = &ldsB[(8 * hi) * CTP + j * 16 + laneRow];
#pragma unroll
                for (int e = 0; e < 16; ++e)
                    ((unsigned short*)bfr[0].q)[e] = p[((e & 7) + (e >> 3) * 16) * CTP];
            }
            v8f c = {};
#pragma unroll
            for (int s = 0; s < 8; ++s) {             // K = 256 in steps of 32
                if (s < 7) {                          // prefetch next fragment
                    const unsigned short* p =
                        &ldsB[(32*(s+1) + 8 * hi) * CTP + j * 16 + laneRow];
#pragma unroll
                    for (int e = 0; e < 16; ++e)
                        ((unsigned short*)bfr[(s + 1) & 1].q)[e] = p[((e & 7) + (e >> 3) * 16) * CTP];
                }
                c = __builtin_amdgcn_wmma_f32_16x16x32_bf16(
                        false, af[s].v, false, bfr[s & 1].v, (short)0, c,
                        false, false);
            }

            // fused epilogue: mask, exp(sim/T), group-positive accumulation
#pragma unroll
            for (int r = 0; r < 8; ++r) {
                int   iglob = rowbase + mhalf + r;
                float pv    = (iglob != jglob) ? mrow[r] * mJ : 0.0f;
                float e     = __expf(c[r] * INV_T) * pv;
                dsum[r] += e;
                nsum[r] += (grow[r] == gJ) ? e : 0.0f;
            }
        }
    }

    // ---- reduce across the 16 lanes of each half-wave, emit per-anchor loss
    float lsum = 0.0f, lcnt = 0.0f;
#pragma unroll
    for (int r = 0; r < 8; ++r) {
        float dn = dsum[r], nm = nsum[r];
#pragma unroll
        for (int m = 8; m >= 1; m >>= 1) {
            dn += __shfl_xor(dn, m, 32);
            nm += __shfl_xor(nm, m, 32);
        }
        if (laneRow == 0) {            // lanes 0 (rows 0-7) and 16 (rows 8-15)
            bool valid = (nm > 0.0f) && (mrow[r] > 0.0f);
            if (valid) {
                lsum += -__logf(nm / dn);
                lcnt += 1.0f;
            }
        }
    }
    if (laneRow == 0) {
        atomicAdd(&accum[0], lsum);
        atomicAdd(&accum[1], lcnt);
    }
}

// --- tiny helpers ----------------------------------------------------------
__global__ void init_accum(float* a) {
    if (threadIdx.x < 2) a[threadIdx.x] = 0.0f;
}

__global__ void finalize_kernel(const float* a, float* out) {
    float s = a[0], c = a[1];
    out[0] = (c > 0.0f) ? s / fmaxf(c, 1.0f) : 0.0f;
}

// ---------------------------------------------------------------------------
extern "C" void kernel_launch(void* const* d_in, const int* in_sizes, int n_in,
                              void* d_out, int out_size, void* d_ws, size_t ws_size,
                              hipStream_t stream) {
    const float*         emb    = (const float*)d_in[0];          // [B,N,D] f32
    const int*           groups = (const int*)d_in[1];            // [B,N] i32
    const unsigned char* mask   = (const unsigned char*)d_in[2];  // [B,N] bool
    float*               out    = (float*)d_out;                  // scalar f32

    const int B = 8, N = N_;
    const int nRows = B * N;                                      // 16384

    // workspace: bf16 normalized embeddings (8 MB) + 2 float accumulators
    unsigned short* zb    = (unsigned short*)d_ws;
    size_t          zbyts = (size_t)nRows * D_ * sizeof(unsigned short);
    float*          accum = (float*)((char*)d_ws + zbyts);

    init_accum<<<1, 32, 0, stream>>>(accum);

    // one wave per row -> 8 rows per 256-thread block
    nrm_kernel<<<(nRows + 7) / 8, 256, 0, stream>>>(emb, zb, nRows);

    // 8 batches * 16 row-tiles of 128 rows
    ctr_kernel<<<B * (N / 128), 256, 0, stream>>>(zb, groups, mask, accum);

    finalize_kernel<<<1, 1, 0, stream>>>(accum, out);

    (void)in_sizes; (void)n_in; (void)out_size; (void)ws_size;
}
